// BasicBlock_2001454760783
// MI455X (gfx1250) — 8-bit matrix operand order repaired and checked by reading, compiled, not run on hardware
//
#include <hip/hip_runtime.h>
#include <stdint.h>

typedef __attribute__((ext_vector_type(8))) int v8i;

#define NB   32
#define CIN  128
#define COUT 128
#define HH   56
#define WW   56
#define HW   (HH*WW)
#define HP   58
#define WPD  58
#define KCH  18
#define CNT_D 100352.0

#define OFF_S      0u
#define SZ_S       (NB*COUT*HW*2)
#define OFF_ACT    (OFF_S + SZ_S)
#define SZ_ACT     (NB*HP*WPD*CIN + 8192)
#define OFF_WPACK  (OFF_ACT + SZ_ACT)
#define SZ_WPACK   (KCH*8*32*32)
#define OFF_WSCALE (OFF_WPACK + SZ_WPACK)
#define OFF_CHSUM  (OFF_WSCALE + 512)
#define OFF_CHSQ   (OFF_CHSUM + 512)
#define OFF_AF     (OFF_CHSQ + 1024)
#define OFF_BF     (OFF_AF + 512)

__global__ __launch_bounds__(256) void binarize_k(const float* __restrict__ x,
                                                  const float* __restrict__ bias0,
                                                  int8_t* __restrict__ act8) {
  size_t idx = (size_t)blockIdx.x * 256 + threadIdx.x;
  int wp = (int)(idx % WPD);
  size_t r = idx / WPD;
  int c  = (int)(r & 127);
  r >>= 7;
  int hp = (int)(r % HP);
  int n  = (int)(r / HP);
  int8_t v = 0;
  if (hp >= 1 && hp <= HH && wp >= 1 && wp <= WW) {
    float t = x[(((size_t)n*CIN + c)*HH + (hp-1))*WW + (wp-1)] + bias0[c];
    v = (int8_t)((t > 0.f) - (t < 0.f));
  }
  act8[(((size_t)n*HP + hp)*WPD + wp)*CIN + c] = v;
}

__global__ void wscale_k(const float* __restrict__ w, float* __restrict__ wscale) {
  int o = threadIdx.x;
  if (o < COUT) {
    float s = 0.f;
    for (int k = 0; k < CIN*9; ++k) s += fabsf(w[(size_t)o*CIN*9 + k]);
    wscale[o] = s / (float)(CIN*9);
  }
}

__global__ void wpack_k(const float* __restrict__ w, uint32_t* __restrict__ wpack) {
  int jt   = blockIdx.x;
  int lane = threadIdx.x;
  int j = jt >> 3, t = jt & 7;
  int hi = lane >> 4, nn = lane & 15;
  int tap = j >> 1;
  int cb  = (j & 1) << 6;
  int o   = t*16 + nn;
  for (int v = 0; v < 8; ++v) {
    int kb = ((v >> 1) << 4) + (hi << 3) + ((v & 1) << 2);
    uint32_t d = 0;
    for (int b = 0; b < 4; ++b) {
      int c = cb + kb + b;
      float wv = w[((size_t)o*CIN + c)*9 + tap];
      int8_t sv = (int8_t)((wv > 0.f) - (wv < 0.f));
      d |= (uint32_t)(uint8_t)sv << (8*b);
    }
    wpack[((size_t)jt*32 + lane)*8 + v] = d;
  }
}

__global__ __launch_bounds__(256) void conv_wmma_k(const int8_t* __restrict__ act8,
                                                   const uint32_t* __restrict__ wpack,
                                                   short* __restrict__ sout,
                                                   int* __restrict__ chsum,
                                                   unsigned long long* __restrict__ chsq) {
  int wave = blockIdx.x * 8 + (threadIdx.x >> 5);
  int lane = threadIdx.x & 31;
  int i  = wave & 3;
  int h  = (wave >> 2) % HH;
  int n  = wave / (4*HH);
  int w0 = i << 4;
  int hi = lane >> 4;
  int mn = lane & 15;

  v8i acc[8];
  const v8i vz = {0,0,0,0,0,0,0,0};
#pragma unroll
  for (int t = 0; t < 8; ++t) acc[t] = vz;

#pragma unroll 1
  for (int j = 0; j < KCH; ++j) {
    int tap = j >> 1;
    int cb  = (j & 1) << 6;
    int hp  = h + tap/3;
    int wp  = w0 + mn + tap%3;
    const int8_t* arow = act8 + ((((size_t)n*HP + hp)*WPD + wp)*CIN) + cb + hi*8;
    int2 p0 = *(const int2*)(arow);
    int2 p1 = *(const int2*)(arow + 16);
    int2 p2 = *(const int2*)(arow + 32);
    int2 p3 = *(const int2*)(arow + 48);
    v8i a;
    a[0]=p0.x; a[1]=p0.y; a[2]=p1.x; a[3]=p1.y;
    a[4]=p2.x; a[5]=p2.y; a[6]=p3.x; a[7]=p3.y;

    const uint32_t* wb = wpack + ((size_t)(j*8)*32 + lane)*8;
#pragma unroll
    for (int t = 0; t < 8; ++t) {
      const int4* bp = (const int4*)(wb + (size_t)t*256);
      int4 b0 = bp[0], b1 = bp[1];
      v8i b;
      b[0]=b0.x; b[1]=b0.y; b[2]=b0.z; b[3]=b0.w;
      b[4]=b1.x; b[5]=b1.y; b[6]=b1.z; b[7]=b1.w;
      acc[t] = __builtin_amdgcn_wmma_i32_16x16x64_iu8(true, a, true, b, acc[t],
                                                      false, false);
    }
  }

  int wlim = WW - w0;
#pragma unroll 1
  for (int t = 0; t < 8; ++t) {
    int o = t*16 + mn;
    int ps = 0, pq = 0;
#pragma unroll
    for (int r = 0; r < 8; ++r) {
      int M = r + 8*hi;
      int v = acc[t][r];
      if (M < wlim) {
        sout[(((size_t)n*COUT + o)*HH + h)*WW + (w0 + M)] = (short)v;
        ps += v;
        pq += v*v;
      }
    }
    ps += __shfl_xor(ps, 16);
    pq += __shfl_xor(pq, 16);
    if (lane < 16) {
      atomicAdd(&chsum[t*16 + lane], ps);
      atomicAdd(&chsq[t*16 + lane], (unsigned long long)(unsigned)pq);
    }
  }
}

__global__ void finalize_k(const int* __restrict__ chsum,
                           const unsigned long long* __restrict__ chsq,
                           const float* __restrict__ wscale,
                           const float* __restrict__ gamma,
                           const float* __restrict__ beta,
                           const float* __restrict__ bias1,
                           float* __restrict__ Af, float* __restrict__ Bf) {
  int c = threadIdx.x;
  if (c < COUT) {
    double ms = (double)chsum[c] / CNT_D;
    double vs = (double)chsq[c] / CNT_D - ms*ms;
    float sc  = wscale[c];
    float var = (float)vs * sc * sc;
    float mean = (float)ms * sc;
    float inv = rsqrtf(var + 1e-5f);
    float g = gamma[c];
    Af[c] = g * inv * sc;
    Bf[c] = beta[c] - mean * g * inv + bias1[c];
  }
}

__global__ __launch_bounds__(256) void apply_k(const short* __restrict__ s,
                                               const float* __restrict__ x,
                                               const float* __restrict__ Af,
                                               const float* __restrict__ Bf,
                                               const float* __restrict__ alpha,
                                               const float* __restrict__ bias2,
                                               float* __restrict__ out) {
  size_t g = (size_t)blockIdx.x * 256 + threadIdx.x;
  int c = (int)((g / (HW/4)) & 127);
  size_t idx = g * 4;
  short4 sv = *(const short4*)(s + idx);
  float4 xv = *(const float4*)(x + idx);
  float A  = Af[c];
  float Bc = Bf[c];
  float al = alpha[c];
  float b2 = bias2[c];
  float4 o;
  float t0 = A * (float)sv.x + Bc + xv.x;
  float t1 = A * (float)sv.y + Bc + xv.y;
  float t2 = A * (float)sv.z + Bc + xv.z;
  float t3 = A * (float)sv.w + Bc + xv.w;
  o.x = ((t0 >= 0.f) ? t0 : al * t0) + b2;
  o.y = ((t1 >= 0.f) ? t1 : al * t1) + b2;
  o.z = ((t2 >= 0.f) ? t2 : al * t2) + b2;
  o.w = ((t3 >= 0.f) ? t3 : al * t3) + b2;
  *(float4*)(out + idx) = o;
}

extern "C" void kernel_launch(void* const* d_in, const int* in_sizes, int n_in,
                              void* d_out, int out_size, void* d_ws, size_t ws_size,
                              hipStream_t stream) {
  const float* x     = (const float*)d_in[0];
  const float* bias0 = (const float*)d_in[1];
  const float* w     = (const float*)d_in[2];
  const float* gamma = (const float*)d_in[3];
  const float* beta  = (const float*)d_in[4];
  const float* bias1 = (const float*)d_in[5];
  const float* alpha = (const float*)d_in[6];
  const float* bias2 = (const float*)d_in[7];
  float* out = (float*)d_out;

  char* ws = (char*)d_ws;
  short*          sbuf   = (short*)(ws + OFF_S);
  int8_t*         act8   = (int8_t*)(ws + OFF_ACT);
  uint32_t*       wpack  = (uint32_t*)(ws + OFF_WPACK);
  float*          wscale = (float*)(ws + OFF_WSCALE);
  int*            chsum  = (int*)(ws + OFF_CHSUM);
  unsigned long long* chsq = (unsigned long long*)(ws + OFF_CHSQ);
  float*          Af     = (float*)(ws + OFF_AF);
  float*          Bf     = (float*)(ws + OFF_BF);

  hipMemsetAsync(ws + OFF_CHSUM, 0, 512 + 1024, stream);

  binarize_k <<<53824, 256, 0, stream>>>(x, bias0, act8);
  wscale_k   <<<1, 128, 0, stream>>>(w, wscale);
  wpack_k    <<<144, 32, 0, stream>>>(w, wpack);
  conv_wmma_k<<<896, 256, 0, stream>>>(act8, wpack, sbuf, chsum, chsq);
  finalize_k <<<1, 128, 0, stream>>>(chsum, chsq, wscale, gamma, beta, bias1, Af, Bf);
  apply_k    <<<12544, 256, 0, stream>>>(sbuf, x, Af, Bf, alpha, bias2, out);
}
